// SparseCrossAttnAdapter_67207648247992
// MI455X (gfx1250) — matrix operand order repaired and checked by reading, not compiled, not run on hardware
//
#include <hip/hip_runtime.h>
#include <hip/hip_bf16.h>
#include <math.h>

// ---------------------------------------------------------------------------
// Types for WMMA fragments (CDNA5 gfx1250, wave32)
// ---------------------------------------------------------------------------
typedef __attribute__((ext_vector_type(16))) __bf16        v16bf;
typedef __attribute__((ext_vector_type(8)))  float         v8f;
typedef __attribute__((ext_vector_type(4)))  unsigned int  v4u;

union Frag {
    v16bf bf;
    v4u   q[2];
    unsigned short us[16];
};

__device__ __forceinline__ unsigned short f2bf(float f) {
    unsigned int u = __float_as_uint(f);
    unsigned int r = u + 0x7fffu + ((u >> 16) & 1u);   // round-to-nearest-even
    return (unsigned short)(r >> 16);
}
__device__ __forceinline__ float bf2f(unsigned short u) {
    return __uint_as_float(((unsigned int)u) << 16);
}
__device__ __forceinline__ v8f v8f_zero() {
    v8f z = {0.f, 0.f, 0.f, 0.f, 0.f, 0.f, 0.f, 0.f};
    return z;
}

#define HID   1024
#define NH    16
#define HD    64
#define BSZ   2
#define LI    4096
#define LC    1024

// ---------------------------------------------------------------------------
// 1) Row-wise RMS norm (1024 cols) + cast fp32 -> bf16
// ---------------------------------------------------------------------------
__global__ __launch_bounds__(256) void rmsnorm_cast_kernel(
    const float* __restrict__ x, const float* __restrict__ w,
    unsigned short* __restrict__ out)
{
    const int row = blockIdx.x;
    const int t   = threadIdx.x;
    const float* xr = x + (size_t)row * HID;

    const float4 v = ((const float4*)xr)[t];
    float s = v.x * v.x + v.y * v.y + v.z * v.z + v.w * v.w;

    __shared__ float red[256];
    red[t] = s;
    __syncthreads();
    for (int off = 128; off > 0; off >>= 1) {
        if (t < off) red[t] += red[t + off];
        __syncthreads();
    }
    const float inv = rsqrtf(red[0] * (1.0f / HID) + 1e-6f);

    const float* wp = w + t * 4;
    unsigned int lo = f2bf(v.x * inv * wp[0]) | ((unsigned int)f2bf(v.y * inv * wp[1]) << 16);
    unsigned int hi = f2bf(v.z * inv * wp[2]) | ((unsigned int)f2bf(v.w * inv * wp[3]) << 16);
    unsigned int* op = (unsigned int*)(out + (size_t)row * HID);
    op[t * 2 + 0] = lo;
    op[t * 2 + 1] = hi;
}

// ---------------------------------------------------------------------------
// 2) Elementwise fp32 -> bf16 cast (weights)
// ---------------------------------------------------------------------------
__global__ __launch_bounds__(256) void cvt_bf16_kernel(
    const float* __restrict__ in, unsigned short* __restrict__ out)
{
    const size_t idx = (size_t)blockIdx.x * 256 + threadIdx.x;   // one float4 each
    const float4 v = ((const float4*)in)[idx];
    unsigned int lo = f2bf(v.x) | ((unsigned int)f2bf(v.y) << 16);
    unsigned int hi = f2bf(v.z) | ((unsigned int)f2bf(v.w) << 16);
    ((unsigned int*)out)[idx * 2 + 0] = lo;
    ((unsigned int*)out)[idx * 2 + 1] = hi;
}

// ---------------------------------------------------------------------------
// 3) GEMM: Y[r,n] = sum_k A[r,k] * W[n,k] + bias[n]   (A,W bf16 row-major)
//    wave = 64 rows x 16 cols; block = 8 waves -> 64 rows x 128 cols
//    mode A (outf==null): store bf16 row-major
//    mode B (outf!=null): outf = residual + alpha * Y   (fp32, final output)
// ---------------------------------------------------------------------------
__global__ __launch_bounds__(256) void gemm_bf16_kernel(
    const unsigned short* __restrict__ A, const unsigned short* __restrict__ W,
    const float* __restrict__ bias,
    unsigned short* __restrict__ outb,
    const float* __restrict__ residual, const float* __restrict__ alphap,
    float* __restrict__ outf)
{
    const int lane = threadIdx.x & 31;
    const int wave = threadIdx.x >> 5;
    const int ll = lane & 15;
    const int lh = lane >> 4;

    const int rowBase = blockIdx.x * 64;
    const int colBase = blockIdx.y * 128 + wave * 16;

    v8f acc[4];
#pragma unroll
    for (int t = 0; t < 4; ++t) acc[t] = v8f_zero();

    const unsigned short* wrow = W + (size_t)(colBase + ll) * HID;

    for (int kb = 0; kb < HID; kb += 32) {
        Frag bfrag;
        const unsigned short* bp = wrow + kb + lh * 8;
        bfrag.q[0] = *(const v4u*)(bp);
        bfrag.q[1] = *(const v4u*)(bp + 16);
#pragma unroll
        for (int t = 0; t < 4; ++t) {
            const unsigned short* ap =
                A + (size_t)(rowBase + t * 16 + ll) * HID + kb + lh * 8;
            Frag afrag;
            afrag.q[0] = *(const v4u*)(ap);
            afrag.q[1] = *(const v4u*)(ap + 16);
            acc[t] = __builtin_amdgcn_wmma_f32_16x16x32_bf16(
                false, afrag.bf, false, bfrag.bf, (short)0, acc[t], false, false);
        }
    }

    const float bval = bias[colBase + ll];
    if (outf) {
        const float alpha = alphap[0];
#pragma unroll
        for (int t = 0; t < 4; ++t) {
            const int row0 = rowBase + t * 16 + lh * 8;
#pragma unroll
            for (int r = 0; r < 8; ++r) {
                const size_t idx = (size_t)(row0 + r) * HID + colBase + ll;
                outf[idx] = residual[idx] + alpha * (acc[t][r] + bval);
            }
        }
    } else {
#pragma unroll
        for (int t = 0; t < 4; ++t) {
            const int row0 = rowBase + t * 16 + lh * 8;
#pragma unroll
            for (int r = 0; r < 8; ++r) {
                const size_t idx = (size_t)(row0 + r) * HID + colBase + ll;
                outb[idx] = f2bf(acc[t][r] + bval);
            }
        }
    }
}

// ---------------------------------------------------------------------------
// 4) Per-head RMS norm + RoPE + optional 1/sqrt(d) scale; layout -> [b,h,t,d]
//    wave = one (token, head); lane owns elems 2*lane, 2*lane+1 (RoPE-local)
// ---------------------------------------------------------------------------
__global__ __launch_bounds__(256) void head_rope_kernel(
    const unsigned short* __restrict__ raw, const float* __restrict__ hw,
    const float* __restrict__ cosT, const float* __restrict__ sinT,
    unsigned short* __restrict__ outh, int L, float scale)
{
    const int lane = threadIdx.x & 31;
    const int wave = threadIdx.x >> 5;
    const long gidx = (long)blockIdx.x * 8 + wave;   // over B*L*NH
    const int h = (int)(gidx % NH);
    const long bt = gidx / NH;
    const int t = (int)(bt % L);
    const int b = (int)(bt / L);

    const unsigned short* src =
        raw + ((size_t)(b * L + t)) * HID + h * HD + lane * 2;
    const unsigned int pr = *(const unsigned int*)src;
    const float x0 = bf2f((unsigned short)(pr & 0xffffu));
    const float x1 = bf2f((unsigned short)(pr >> 16));

    float ss = x0 * x0 + x1 * x1;
#pragma unroll
    for (int m = 16; m > 0; m >>= 1) ss += __shfl_xor(ss, m, 32);
    const float inv = rsqrtf(ss * (1.0f / HD) + 1e-6f);

    const int d = lane * 2;
    const float n0 = x0 * inv * hw[d];
    const float n1 = x1 * inv * hw[d + 1];
    const float c0 = cosT[(size_t)t * HD + d];
    const float c1 = cosT[(size_t)t * HD + d + 1];
    const float s0 = sinT[(size_t)t * HD + d];
    const float s1 = sinT[(size_t)t * HD + d + 1];
    const float r0 = n0 * c0 - n1 * s0;
    const float r1 = n1 * c1 + n0 * s1;

    const size_t o = ((((size_t)b * NH + h) * L + t) * HD) + d;
    unsigned int packed =
        f2bf(r0 * scale) | ((unsigned int)f2bf(r1 * scale) << 16);
    *(unsigned int*)(outh + o) = packed;
}

// ---------------------------------------------------------------------------
// 5) V transpose: vt[b][h][d][t] = vraw[(b*L+t)*1024 + h*64 + d]
// ---------------------------------------------------------------------------
__global__ __launch_bounds__(256) void v_transpose_kernel(
    const unsigned short* __restrict__ vraw, unsigned short* __restrict__ vt, int L)
{
    long i = (long)blockIdx.x * 256 + threadIdx.x;    // over B*NH*HD*L
    const int t = (int)(i % L); long rest = i / L;
    const int d = (int)(rest % HD); rest /= HD;
    const int h = (int)(rest % NH);
    const int b = (int)(rest / NH);
    vt[i] = vraw[((size_t)(b * L + t)) * HID + h * HD + d];
}

// ---------------------------------------------------------------------------
// 6) Flash-style attention. block = (b,h) x 128 queries; wave = 16 queries.
//    Per 32-key step: 4 WMMAs for S, online softmax, P via LDS relayout,
//    4 WMMAs for P*V. Output written token-major [b][q][h*64+d] in bf16.
// ---------------------------------------------------------------------------
__global__ __launch_bounds__(256) void attn_kernel(
    const unsigned short* __restrict__ Qh,   // [B][NH][Li][64]
    const unsigned short* __restrict__ Kh,   // [B][NH][Lc][64]
    const unsigned short* __restrict__ Vt,   // [B][NH][64][Lc]
    unsigned short* __restrict__ AO)         // [B][Li][1024]
{
    __shared__ __align__(16) unsigned short pbuf[8][16][32];

    const int lane = threadIdx.x & 31;
    const int wave = threadIdx.x >> 5;
    const int ll = lane & 15;
    const int lh = lane >> 4;

    const int qBase = blockIdx.x * 128 + wave * 16;
    const int h = blockIdx.y & (NH - 1);
    const int b = blockIdx.y >> 4;

    const unsigned short* Qp = Qh + ((size_t)(b * NH + h)) * LI * HD;
    const unsigned short* Kp = Kh + ((size_t)(b * NH + h)) * LC * HD;
    const unsigned short* Vp = Vt + ((size_t)(b * NH + h)) * HD * LC;

    // Q fragments: A 16x32 for d=[0,32) and d=[32,64)
    Frag qf[2];
    {
        const unsigned short* qp = Qp + (size_t)(qBase + ll) * HD + lh * 8;
        qf[0].q[0] = *(const v4u*)(qp);
        qf[0].q[1] = *(const v4u*)(qp + 16);
        qf[1].q[0] = *(const v4u*)(qp + 32);
        qf[1].q[1] = *(const v4u*)(qp + 48);
    }

    v8f acc[4];
#pragma unroll
    for (int t = 0; t < 4; ++t) acc[t] = v8f_zero();
    float mrow[8], srow[8];
#pragma unroll
    for (int r = 0; r < 8; ++r) { mrow[r] = -1e30f; srow[r] = 0.f; }

    for (int kb = 0; kb < LC; kb += 32) {
        // ---- scores: two 16x16 tiles (keys kb..kb+15 and kb+16..kb+31) ----
        v8f s0 = v8f_zero(), s1 = v8f_zero();
        {
            const unsigned short* kp = Kp + (size_t)(kb + ll) * HD + lh * 8;
            Frag bk;
            bk.q[0] = *(const v4u*)(kp);      bk.q[1] = *(const v4u*)(kp + 16);
            s0 = __builtin_amdgcn_wmma_f32_16x16x32_bf16(false, qf[0].bf, false, bk.bf, (short)0, s0, false, false);
            bk.q[0] = *(const v4u*)(kp + 32); bk.q[1] = *(const v4u*)(kp + 48);
            s0 = __builtin_amdgcn_wmma_f32_16x16x32_bf16(false, qf[1].bf, false, bk.bf, (short)0, s0, false, false);
        }
        {
            const unsigned short* kp = Kp + (size_t)(kb + 16 + ll) * HD + lh * 8;
            Frag bk;
            bk.q[0] = *(const v4u*)(kp);      bk.q[1] = *(const v4u*)(kp + 16);
            s1 = __builtin_amdgcn_wmma_f32_16x16x32_bf16(false, qf[0].bf, false, bk.bf, (short)0, s1, false, false);
            bk.q[0] = *(const v4u*)(kp + 32); bk.q[1] = *(const v4u*)(kp + 48);
            s1 = __builtin_amdgcn_wmma_f32_16x16x32_bf16(false, qf[1].bf, false, bk.bf, (short)0, s1, false, false);
        }

        // ---- online softmax update (row stats match C layout) ----
        float p0[8], p1[8], scl[8];
#pragma unroll
        for (int r = 0; r < 8; ++r) {
            float mx = fmaxf(s0[r], s1[r]);
#pragma unroll
            for (int m = 8; m > 0; m >>= 1) mx = fmaxf(mx, __shfl_xor(mx, m, 16));
            const float nm = fmaxf(mrow[r], mx);
            scl[r] = __expf(mrow[r] - nm);
            mrow[r] = nm;
            p0[r] = __expf(s0[r] - nm);
            p1[r] = __expf(s1[r] - nm);
            float sm = p0[r] + p1[r];
#pragma unroll
            for (int m = 8; m > 0; m >>= 1) sm += __shfl_xor(sm, m, 16);
            srow[r] = srow[r] * scl[r] + sm;
        }
#pragma unroll
        for (int t = 0; t < 4; ++t)
#pragma unroll
            for (int r = 0; r < 8; ++r) acc[t][r] *= scl[r];

        // ---- P (C layout) -> LDS -> A fragment (16x32) ----
        const int prow = lh * 8;
#pragma unroll
        for (int r = 0; r < 8; ++r) {
            pbuf[wave][prow + r][ll]      = f2bf(p0[r]);
            pbuf[wave][prow + r][ll + 16] = f2bf(p1[r]);
        }
        asm volatile("s_wait_dscnt 0" ::: "memory");
        Frag pf;
        {
            const unsigned short* pp = &pbuf[wave][ll][lh * 8];
            pf.q[0] = *(const v4u*)(pp);
            pf.q[1] = *(const v4u*)(pp + 16);
        }
        asm volatile("s_wait_dscnt 0" ::: "memory");

        // ---- O += P * V : 4 column tiles of d ----
#pragma unroll
        for (int nt = 0; nt < 4; ++nt) {
            const unsigned short* vp =
                Vp + (size_t)(nt * 16 + ll) * LC + kb + lh * 8;
            Frag bv;
            bv.q[0] = *(const v4u*)(vp);
            bv.q[1] = *(const v4u*)(vp + 16);
            acc[nt] = __builtin_amdgcn_wmma_f32_16x16x32_bf16(
                false, pf.bf, false, bv.bf, (short)0, acc[nt], false, false);
        }
    }

    // ---- finalize: divide by row sum, store bf16 token-major ----
    const size_t aobase = (size_t)b * LI * HID;
#pragma unroll
    for (int nt = 0; nt < 4; ++nt) {
#pragma unroll
        for (int r = 0; r < 8; ++r) {
            const int q = qBase + lh * 8 + r;
            AO[aobase + (size_t)q * HID + h * HD + nt * 16 + ll] =
                f2bf(acc[nt][r] / srow[r]);
        }
    }
}

// ---------------------------------------------------------------------------
// Launcher
// ---------------------------------------------------------------------------
extern "C" void kernel_launch(void* const* d_in, const int* in_sizes, int n_in,
                              void* d_out, int out_size, void* d_ws, size_t ws_size,
                              hipStream_t stream)
{
    const float* img  = (const float*)d_in[0];
    const float* cnd  = (const float*)d_in[1];
    const float* qnw  = (const float*)d_in[2];
    const float* kvnw = (const float*)d_in[3];
    const float* qhw  = (const float*)d_in[4];
    const float* khw  = (const float*)d_in[5];
    const float* qw   = (const float*)d_in[6];
    const float* qb   = (const float*)d_in[7];
    const float* kw   = (const float*)d_in[8];
    const float* kb_  = (const float*)d_in[9];
    const float* vw   = (const float*)d_in[10];
    const float* vb   = (const float*)d_in[11];
    const float* ow   = (const float*)d_in[12];
    const float* ob   = (const float*)d_in[13];
    const float* alpha= (const float*)d_in[14];
    const float* icos = (const float*)d_in[15];
    const float* isin = (const float*)d_in[16];
    const float* ccos = (const float*)d_in[17];
    const float* csin = (const float*)d_in[18];
    float* out = (float*)d_out;

    char* ws = (char*)d_ws;
    // byte offsets into workspace (bf16 buffers); XN reused for QH, KVN for KH,
    // QRAW reused for attention output (all hazard-free: strictly sequential).
    unsigned short* XN   = (unsigned short*)(ws + 0);            // 16 MB (later QH)
    unsigned short* KVN  = (unsigned short*)(ws + 16777216);     //  4 MB (later KH)
    unsigned short* QWb  = (unsigned short*)(ws + 20971520);     //  2 MB
    unsigned short* KWb  = (unsigned short*)(ws + 23068672);
    unsigned short* VWb  = (unsigned short*)(ws + 25165824);
    unsigned short* OWb  = (unsigned short*)(ws + 27262976);
    unsigned short* QRAW = (unsigned short*)(ws + 29360128);     // 16 MB (later AO)
    unsigned short* KRAW = (unsigned short*)(ws + 46137344);     //  4 MB
    unsigned short* VRAW = (unsigned short*)(ws + 50331648);     //  4 MB
    unsigned short* VT   = (unsigned short*)(ws + 54525952);     //  4 MB
    unsigned short* QH = XN;
    unsigned short* KH = KVN;
    unsigned short* AO = QRAW;

    // 1) RMS norms -> bf16
    rmsnorm_cast_kernel<<<BSZ * LI, 256, 0, stream>>>(img, qnw, XN);
    rmsnorm_cast_kernel<<<BSZ * LC, 256, 0, stream>>>(cnd, kvnw, KVN);

    // 2) weight casts (1M elems each, 4 per thread)
    cvt_bf16_kernel<<<1024, 256, 0, stream>>>(qw, QWb);
    cvt_bf16_kernel<<<1024, 256, 0, stream>>>(kw, KWb);
    cvt_bf16_kernel<<<1024, 256, 0, stream>>>(vw, VWb);
    cvt_bf16_kernel<<<1024, 256, 0, stream>>>(ow, OWb);

    // 3) Q/K/V projections (WMMA)
    gemm_bf16_kernel<<<dim3(BSZ * LI / 64, HID / 128), 256, 0, stream>>>(
        XN, QWb, qb, QRAW, nullptr, nullptr, nullptr);
    gemm_bf16_kernel<<<dim3(BSZ * LC / 64, HID / 128), 256, 0, stream>>>(
        KVN, KWb, kb_, KRAW, nullptr, nullptr, nullptr);
    gemm_bf16_kernel<<<dim3(BSZ * LC / 64, HID / 128), 256, 0, stream>>>(
        KVN, VWb, vb, VRAW, nullptr, nullptr, nullptr);

    // 4) head norm + RoPE (1/sqrt(64) folded into Q), V transpose
    head_rope_kernel<<<BSZ * LI * NH / 8, 256, 0, stream>>>(
        QRAW, qhw, icos, isin, QH, LI, 0.125f);
    head_rope_kernel<<<BSZ * LC * NH / 8, 256, 0, stream>>>(
        KRAW, khw, ccos, csin, KH, LC, 1.0f);
    v_transpose_kernel<<<BSZ * NH * HD * LC / 256, 256, 0, stream>>>(VRAW, VT, LC);

    // 5) attention (WMMA flash)
    attn_kernel<<<dim3(LI / 128, BSZ * NH), 256, 0, stream>>>(QH, KH, VT, AO);

    // 6) O projection + residual (fp32 output)
    gemm_bf16_kernel<<<dim3(BSZ * LI / 64, HID / 128), 256, 0, stream>>>(
        AO, OWb, ob, nullptr, img, alpha, out);
}
